// GPT_Compressive_85779086835946
// MI455X (gfx1250) — matrix operand order repaired and checked by reading, not compiled, not run on hardware
//
#include <hip/hip_runtime.h>

// ---------------------------------------------------------------------------
// Model constants (must match the reference)
// ---------------------------------------------------------------------------
#define LAYERS 6
#define HEADS  8
#define HID    512
#define FFW    2048
#define VOC    32000
#define SEQL   2048
#define NSL    512      // tokens per block
#define NML    512      // recent memory
#define NCML   512      // compressed memory
#define EMB    128
#define NFCL   128      // NS / C_FACTOR
#define NMEML  1024     // NM + NCM
#define NBLK   4
#define HD     64
#define BB     2        // batch

// ---------------------------------------------------------------------------
// WMMA types (CDNA5 / gfx1250, wave32)
// ---------------------------------------------------------------------------
typedef __attribute__((ext_vector_type(16))) __bf16 bf16x16;
typedef __attribute__((ext_vector_type(8)))  __bf16 bf16x8;
typedef __attribute__((ext_vector_type(8)))  float  f32x8;
typedef __attribute__((ext_vector_type(4)))  float  f32x4;
typedef __attribute__((ext_vector_type(4)))  int    v4i;

union Frag16 { bf16x16 v; bf16x8 h[2]; };

// GEMM tiling
#define TM 128
#define TN 128
#define TK 32
#define LDSK  40   // bf16 LDS row stride: 80B, multiple of 16B
#define LDSKF 36   // fp32 LDS row stride: 144B, multiple of 16B

// ---------------------------------------------------------------------------
// Async global->LDS copy (CDNA5): 16 bytes per lane, tracked by ASYNCcnt.
// Builtin prototype: (v4i addrspace(1)*, v4i addrspace(3)*, imm offset, imm cpol)
// ---------------------------------------------------------------------------
typedef __attribute__((address_space(1))) v4i g_v4i;
typedef __attribute__((address_space(3))) v4i l_v4i;

__device__ __forceinline__ void gl2lds_b128(const void* g, void* l) {
#if __has_builtin(__builtin_amdgcn_global_load_async_to_lds_b128)
    __builtin_amdgcn_global_load_async_to_lds_b128(
        (g_v4i*)(void*)g, (l_v4i*)l, 0, 0);
#else
    unsigned loff = (unsigned)(unsigned long)(l_v4i*)l;
    asm volatile("global_load_async_to_lds_b128 %0, %1, off"
                 :: "v"(loff), "v"(g) : "memory");
#endif
}

__device__ __forceinline__ void wait_async0() {
#if __has_builtin(__builtin_amdgcn_s_wait_asynccnt)
    __builtin_amdgcn_s_wait_asynccnt(0);
#else
    asm volatile("s_wait_asynccnt 0x0" ::: "memory");
#endif
}

// ---------------------------------------------------------------------------
// Weight-GEMM:  C[z] = relu( alpha * A[z] x Wt^T + bias )
//   A  : fp32, M x K row-major (M % 128 == 0, K % 32 == 0)
//   Wt : bf16, N x K row-major (pre-transposed weight; N % 128 == 0)
// Async-copied tiles, double-buffered LDS, bf16 WMMA, fp32 accumulate.
// Batch: off = (z/cnt)*Outer + (z%cnt)*Inner  (elements).
// ---------------------------------------------------------------------------
__global__ __launch_bounds__(256) void gemm_wt_k(
    const float* __restrict__ A, const __bf16* __restrict__ Wt,
    const float* __restrict__ bias, float* __restrict__ C,
    int M, int N, int K, int lda, int ldc,
    long aO, long aI, long cO, long cI, int cnt,
    int relu, float alpha)
{
    __shared__ float  As[2][TM * LDSKF];   // fp32 A tiles (double buffered)
    __shared__ __bf16 Bs[2][TN * LDSK];    // bf16 Wt tiles

    const int z  = blockIdx.z;
    const long zo = z / cnt, zi = z % cnt;
    A += zo * aO + zi * aI;
    C += zo * cO + zi * cI;

    const int m0 = blockIdx.y * TM;
    const int n0 = blockIdx.x * TN;
    const int tid  = threadIdx.x;
    const int lane = tid & 31;
    const int w    = tid >> 5;
    const int wm   = w >> 1;            // 0..3
    const int wn   = w & 1;             // 0..1
    const int laneM  = lane & 15;
    const int laneHi = lane >> 4;

    f32x8 acc[2][4];
    #pragma unroll
    for (int i = 0; i < 2; ++i)
        #pragma unroll
        for (int j = 0; j < 4; ++j)
            #pragma unroll
            for (int e = 0; e < 8; ++e) acc[i][j][e] = 0.0f;

    // issue async copies for one K-tile into buffer `buf`
    auto issue = [&](int kb, int buf) {
        // A tile: TM*TK fp32 = 1024 x 16B chunks (4 per thread)
        #pragma unroll
        for (int t = 0; t < 4; ++t) {
            int c  = tid + t * 256;
            int mm = c >> 3;            // 8 chunks per row
            int kk = (c & 7) * 4;
            gl2lds_b128(A + (long)(m0 + mm) * lda + kb + kk,
                        &As[buf][mm * LDSKF + kk]);
        }
        // Wt tile: TN*TK bf16 = 512 x 16B chunks (2 per thread)
        #pragma unroll
        for (int t = 0; t < 2; ++t) {
            int c  = tid + t * 256;
            int nn = c >> 2;            // 4 chunks per row
            int kk = (c & 3) * 8;
            gl2lds_b128(Wt + (long)(n0 + nn) * K + kb + kk,
                        &Bs[buf][nn * LDSK + kk]);
        }
    };

    issue(0, 0);
    int cur = 0;
    for (int k0 = 0; k0 < K; k0 += TK, cur ^= 1) {
        wait_async0();
        __syncthreads();
        if (k0 + TK < K) issue(k0 + TK, cur ^ 1);

        // A fragments: fp32 from LDS, convert to bf16 at the register level
        Frag16 a[2], b[4];
        #pragma unroll
        for (int mf = 0; mf < 2; ++mf) {
            int r  = wm * 32 + mf * 16 + laneM;
            int c0 = laneHi * 8;
            const float* p = &As[cur][r * LDSKF];
            f32x4 f0 = *(const f32x4*)(p + c0);
            f32x4 f1 = *(const f32x4*)(p + c0 + 4);
            f32x4 f2 = *(const f32x4*)(p + c0 + 16);
            f32x4 f3 = *(const f32x4*)(p + c0 + 20);
            #pragma unroll
            for (int j = 0; j < 4; ++j) {
                a[mf].v[j]      = (__bf16)f0[j];
                a[mf].v[4 + j]  = (__bf16)f1[j];
                a[mf].v[8 + j]  = (__bf16)f2[j];
                a[mf].v[12 + j] = (__bf16)f3[j];
            }
        }
        // B fragments: bf16 straight from LDS (N-major layout)
        #pragma unroll
        for (int nf = 0; nf < 4; ++nf) {
            int n  = wn * 64 + nf * 16 + laneM;
            int c0 = laneHi * 8;
            b[nf].h[0] = *(const bf16x8*)&Bs[cur][n * LDSK + c0];
            b[nf].h[1] = *(const bf16x8*)&Bs[cur][n * LDSK + c0 + 16];
        }
        #pragma unroll
        for (int mf = 0; mf < 2; ++mf)
            #pragma unroll
            for (int nf = 0; nf < 4; ++nf)
                acc[mf][nf] = __builtin_amdgcn_wmma_f32_16x16x32_bf16(
                    false, a[mf].v, false, b[nf].v,
                    (short)0, acc[mf][nf], false, false);
    }

    // epilogue (no bounds guards: M % 128 == 0, N % 128 == 0)
    #pragma unroll
    for (int mf = 0; mf < 2; ++mf) {
        #pragma unroll
        for (int nf = 0; nf < 4; ++nf) {
            int col = n0 + wn * 64 + nf * 16 + laneM;
            float bv = bias ? bias[col] : 0.0f;
            #pragma unroll
            for (int r = 0; r < 8; ++r) {
                int row = m0 + wm * 32 + mf * 16 + r + laneHi * 8;
                float v = acc[mf][nf][r] * alpha + bv;
                if (relu) v = fmaxf(v, 0.0f);
                C[(long)row * ldc + col] = v;
            }
        }
    }
}

// ---------------------------------------------------------------------------
// Generic batched GEMM (guarded, VGPR staging) for attention/compression
// odd shapes: C[z] = alpha * op(A) x op(B).  fp32 in/out, bf16 WMMA.
// ---------------------------------------------------------------------------
__global__ __launch_bounds__(256) void gemm_k(
    const float* __restrict__ A, const float* __restrict__ Bm,
    float* __restrict__ C,
    int M, int N, int K, int lda, int ldb, int ldc,
    long aO, long aI, long bO, long bI, long cO, long cI, int cnt,
    int tA, int tB, float alpha)
{
    __shared__ __bf16 As[TM * LDSK];
    __shared__ __bf16 Bs[TN * LDSK];

    const int z  = blockIdx.z;
    const long zo = z / cnt, zi = z % cnt;
    A  += zo * aO + zi * aI;
    Bm += zo * bO + zi * bI;
    C  += zo * cO + zi * cI;

    const int m0 = blockIdx.y * TM;
    const int n0 = blockIdx.x * TN;
    const int tid  = threadIdx.x;
    const int lane = tid & 31;
    const int w    = tid >> 5;
    const int wm   = w >> 1;
    const int wn   = w & 1;
    const int laneM  = lane & 15;
    const int laneHi = lane >> 4;

    f32x8 acc[2][4];
    #pragma unroll
    for (int i = 0; i < 2; ++i)
        #pragma unroll
        for (int j = 0; j < 4; ++j)
            #pragma unroll
            for (int e = 0; e < 8; ++e) acc[i][j][e] = 0.0f;

    for (int k0 = 0; k0 < K; k0 += TK) {
        for (int i = tid; i < TM * TK; i += 256) {
            int mm = i >> 5, kk = i & 31;
            int gm = m0 + mm, gk = k0 + kk;
            float v = 0.0f;
            if (gm < M && gk < K)
                v = tA ? A[(long)gk * lda + gm] : A[(long)gm * lda + gk];
            As[mm * LDSK + kk] = (__bf16)v;
        }
        for (int i = tid; i < TN * TK; i += 256) {
            int nn = i >> 5, kk = i & 31;
            int gn = n0 + nn, gk = k0 + kk;
            float v = 0.0f;
            if (gn < N && gk < K)
                v = tB ? Bm[(long)gn * ldb + gk] : Bm[(long)gk * ldb + gn];
            Bs[nn * LDSK + kk] = (__bf16)v;
        }
        __syncthreads();

        Frag16 a[2], b[4];
        #pragma unroll
        for (int mf = 0; mf < 2; ++mf) {
            int r  = wm * 32 + mf * 16 + laneM;
            int c0 = laneHi * 8;
            a[mf].h[0] = *(const bf16x8*)&As[r * LDSK + c0];
            a[mf].h[1] = *(const bf16x8*)&As[r * LDSK + c0 + 16];
        }
        #pragma unroll
        for (int nf = 0; nf < 4; ++nf) {
            int n  = wn * 64 + nf * 16 + laneM;
            int c0 = laneHi * 8;
            b[nf].h[0] = *(const bf16x8*)&Bs[n * LDSK + c0];
            b[nf].h[1] = *(const bf16x8*)&Bs[n * LDSK + c0 + 16];
        }
        #pragma unroll
        for (int mf = 0; mf < 2; ++mf)
            #pragma unroll
            for (int nf = 0; nf < 4; ++nf)
                acc[mf][nf] = __builtin_amdgcn_wmma_f32_16x16x32_bf16(
                    false, a[mf].v, false, b[nf].v,
                    (short)0, acc[mf][nf], false, false);
        __syncthreads();
    }

    #pragma unroll
    for (int mf = 0; mf < 2; ++mf) {
        #pragma unroll
        for (int nf = 0; nf < 4; ++nf) {
            int col = n0 + wn * 64 + nf * 16 + laneM;
            if (col >= N) continue;
            #pragma unroll
            for (int r = 0; r < 8; ++r) {
                int row = m0 + wm * 32 + mf * 16 + r + laneHi * 8;
                if (row >= M) continue;
                C[(long)row * ldc + col] = acc[mf][nf][r] * alpha;
            }
        }
    }
}

// ---------------------------------------------------------------------------
// Weight transpose + fp32 -> bf16:  Wt[n*K + k] = (bf16) W[k*N + n]
// ---------------------------------------------------------------------------
__global__ __launch_bounds__(256) void transpose_bf16_k(
    const float* __restrict__ W, __bf16* __restrict__ Wt, int K, int N)
{
    __shared__ float t[32][33];
    const int k0 = blockIdx.y * 32, n0 = blockIdx.x * 32;
    for (int dy = threadIdx.y; dy < 32; dy += 8) {
        int k = k0 + dy, n = n0 + threadIdx.x;
        t[dy][threadIdx.x] = (k < K && n < N) ? W[(long)k * N + n] : 0.0f;
    }
    __syncthreads();
    for (int dy = threadIdx.y; dy < 32; dy += 8) {
        int n = n0 + dy, k = k0 + threadIdx.x;
        if (n < N && k < K) Wt[(long)n * K + k] = (__bf16)t[threadIdx.x][dy];
    }
}

// ---------------------------------------------------------------------------
// LayerNorm (+ optional residual / positional broadcast)
// ---------------------------------------------------------------------------
__global__ __launch_bounds__(128) void ln_k(
    float* __restrict__ out, const float* __restrict__ x,
    const float* __restrict__ res, const float* __restrict__ pos,
    const float* __restrict__ gamma, const float* __restrict__ beta)
{
    const int row = blockIdx.x;
    const int tid = threadIdx.x;
    const float* xr = x + (long)row * HID;
    __shared__ float sh[128];

    float s = 0.f, s2 = 0.f;
    for (int i = tid; i < HID; i += 128) { float v = xr[i]; s += v; s2 += v * v; }
    sh[tid] = s; __syncthreads();
    for (int o = 64; o > 0; o >>= 1) { if (tid < o) sh[tid] += sh[tid + o]; __syncthreads(); }
    float mean = sh[0] * (1.0f / HID);
    __syncthreads();
    sh[tid] = s2; __syncthreads();
    for (int o = 64; o > 0; o >>= 1) { if (tid < o) sh[tid] += sh[tid + o]; __syncthreads(); }
    float var  = sh[0] * (1.0f / HID) - mean * mean;
    float rstd = rsqrtf(var + 1e-6f);

    const float* rr = res ? res + (long)row * HID : nullptr;
    const float* pp = pos ? pos + (long)(row % NSL) * HID : nullptr;
    float* orow = out + (long)row * HID;
    for (int i = tid; i < HID; i += 128) {
        float v = (xr[i] - mean) * rstd * gamma[i] + beta[i];
        if (rr) v += rr[i];
        if (pp) v += pp[i];
        orow[i] = v;
    }
}

// ---------------------------------------------------------------------------
// Row softmax (optional causal prefix); grid = (NSL, B*heads)
// ---------------------------------------------------------------------------
__global__ __launch_bounds__(256) void softmax_rows_k(
    float* __restrict__ S, int len, int causal)
{
    const int q = blockIdx.x;
    float* p = S + ((long)blockIdx.y * NSL + q) * len;
    const int valid = causal ? (q + 1) : len;
    const int tid = threadIdx.x;
    __shared__ float sh[256];

    float m = -1e30f;
    for (int i = tid; i < valid; i += 256) m = fmaxf(m, p[i]);
    sh[tid] = m; __syncthreads();
    for (int o = 128; o > 0; o >>= 1) { if (tid < o) sh[tid] = fmaxf(sh[tid], sh[tid + o]); __syncthreads(); }
    m = sh[0]; __syncthreads();

    float s = 0.f;
    for (int i = tid; i < valid; i += 256) { float e = __expf(p[i] - m); p[i] = e; s += e; }
    sh[tid] = s; __syncthreads();
    for (int o = 128; o > 0; o >>= 1) { if (tid < o) sh[tid] += sh[tid + o]; __syncthreads(); }
    float inv = 1.0f / sh[0];

    for (int i = tid; i < len; i += 256) p[i] = (i < valid) ? p[i] * inv : 0.0f;
}

// ---------------------------------------------------------------------------
// Softmax over axis 1 of (B, NML, NFCL)
// ---------------------------------------------------------------------------
__global__ __launch_bounds__(128) void softmax_axis1_k(float* __restrict__ A)
{
    float* base = A + (long)blockIdx.y * NML * NFCL + blockIdx.x;
    const int tid = threadIdx.x;
    __shared__ float sh[128];

    float m = -1e30f;
    for (int i = tid; i < NML; i += 128) m = fmaxf(m, base[(long)i * NFCL]);
    sh[tid] = m; __syncthreads();
    for (int o = 64; o > 0; o >>= 1) { if (tid < o) sh[tid] = fmaxf(sh[tid], sh[tid + o]); __syncthreads(); }
    m = sh[0]; __syncthreads();

    float s = 0.f;
    for (int i = tid; i < NML; i += 128) {
        float e = __expf(base[(long)i * NFCL] - m);
        base[(long)i * NFCL] = e; s += e;
    }
    sh[tid] = s; __syncthreads();
    for (int o = 64; o > 0; o >>= 1) { if (tid < o) sh[tid] += sh[tid + o]; __syncthreads(); }
    float inv = 1.0f / sh[0];
    for (int i = tid; i < NML; i += 128) base[(long)i * NFCL] *= inv;
}

// ---------------------------------------------------------------------------
// Small elementwise / data-movement kernels
// ---------------------------------------------------------------------------
__global__ void zero_k(float* __restrict__ p, long n) {
    long i = (long)blockIdx.x * blockDim.x + threadIdx.x;
    for (; i < n; i += (long)gridDim.x * blockDim.x) p[i] = 0.0f;
}

__global__ void add_k(float* __restrict__ o, const float* __restrict__ a,
                      const float* __restrict__ b, long n) {
    long i = (long)blockIdx.x * blockDim.x + threadIdx.x;
    if (i < n) o[i] = a[i] + b[i];
}

__global__ void add_bcast_k(float* __restrict__ o, const float* __restrict__ a,
                            const float* __restrict__ bc, long n, long per) {
    long i = (long)blockIdx.x * blockDim.x + threadIdx.x;
    if (i < n) o[i] = a[i] + bc[i % per];
}

__global__ __launch_bounds__(128) void embed_k(
    float* __restrict__ Ae, const int* __restrict__ ids,
    const float* __restrict__ Wemb, int nbOff)
{
    const int row = blockIdx.x;
    const int b = row / NSL, s = row % NSL;
    const int tok = ids[(long)b * SEQL + nbOff + s];
    const float* src = Wemb + (long)tok * EMB;
    float* dst = Ae + (long)row * EMB;
    for (int i = threadIdx.x; i < EMB; i += 128) dst[i] = src[i];
}

__global__ __launch_bounds__(256) void update_mem_k(
    float* __restrict__ mNew, const float* __restrict__ mOld,
    const float* __restrict__ comp, const float* __restrict__ lay)
{
    const int row = blockIdx.x;                 // 0..NMEML-1
    const int b   = blockIdx.y;
    const float* src;
    if (row < 384)      src = mOld + ((long)b * NMEML + 128 + row) * HID;
    else if (row < 512) src = comp + ((long)b * NFCL + (row - 384)) * HID;
    else                src = lay  + ((long)b * NSL  + (row - 512)) * HID;
    float* dst = mNew + ((long)b * NMEML + row) * HID;
    for (int i = threadIdx.x; i < HID; i += 256) dst[i] = src[i];
}

// ---------------------------------------------------------------------------
// Host-side helpers
// ---------------------------------------------------------------------------
static inline void gemm_w(hipStream_t st, const float* A, const __bf16* Wt,
                          const float* bias, float* C,
                          int M, int N, int K, int lda, int ldc,
                          long aO, long aI, long cO, long cI, int cnt,
                          int relu, float alpha, int nz)
{
    dim3 g(N / TN, M / TM, nz);
    gemm_wt_k<<<g, 256, 0, st>>>(A, Wt, bias, C, M, N, K, lda, ldc,
                                 aO, aI, cO, cI, cnt, relu, alpha);
}

static inline void gemm(hipStream_t st, const float* A, const float* B,
                        float* C, int M, int N, int K, int lda, int ldb, int ldc,
                        long aO, long aI, long bO, long bI, long cO, long cI,
                        int cnt, int tA, int tB, float alpha, int nz)
{
    dim3 g((N + TN - 1) / TN, (M + TM - 1) / TM, nz);
    gemm_k<<<g, 256, 0, st>>>(A, B, C, M, N, K, lda, ldb, ldc,
                              aO, aI, bO, bI, cO, cI, cnt, tA, tB, alpha);
}

// ---------------------------------------------------------------------------
// Orchestration
// ---------------------------------------------------------------------------
extern "C" void kernel_launch(void* const* d_in, const int* in_sizes, int n_in,
                              void* d_out, int out_size, void* d_ws, size_t ws_size,
                              hipStream_t stream)
{
    (void)in_sizes; (void)n_in; (void)out_size; (void)ws_size;

    const int*   ids        = (const int*)  d_in[0];
    const float* W_emb      = (const float*)d_in[1];
    const float* W_dec_lin  = (const float*)d_in[2];
    const float* p_compress = (const float*)d_in[3];
    const float* p_c_linear = (const float*)d_in[4];
    const float* p_decoder  = (const float*)d_in[5];
    const float* p_d_q      = (const float*)d_in[6];
    const float* p_d_k      = (const float*)d_in[7];
    const float* p_d_v      = (const float*)d_in[8];
    const float* p_d_c      = (const float*)d_in[9];
    const float* p_m_q      = (const float*)d_in[10];
    const float* p_m_k      = (const float*)d_in[11];
    const float* p_m_v      = (const float*)d_in[12];
    const float* p_m_c      = (const float*)d_in[13];
    const float* p_d_ff1    = (const float*)d_in[14];
    const float* p_d_ff2    = (const float*)d_in[15];
    const float* p_m_ff1    = (const float*)d_in[16];
    const float* p_m_ff2    = (const float*)d_in[17];
    const float* b_d_ff1    = (const float*)d_in[18];
    const float* b_m_ff1    = (const float*)d_in[19];
    const float* b_d_ff2    = (const float*)d_in[20];
    const float* b_m_ff2    = (const float*)d_in[21];
    const float* d_o_bias   = (const float*)d_in[22];
    const float* d_o_scale  = (const float*)d_in[23];
    const float* b_d_bias_1 = (const float*)d_in[24];
    const float* b_d_bias_2 = (const float*)d_in[25];
    const float* b_i_bias   = (const float*)d_in[26];
    const float* b_m_bias_1 = (const float*)d_in[27];
    const float* b_m_bias_2 = (const float*)d_in[28];
    const float* b_d_scale_1= (const float*)d_in[29];
    const float* b_d_scale_2= (const float*)d_in[30];
    const float* b_i_scale  = (const float*)d_in[31];
    const float* b_m_scale_1= (const float*)d_in[32];
    const float* b_m_scale_2= (const float*)d_in[33];
    const float* x_pos      = (const float*)d_in[34];
    const float* x_mem_emb  = (const float*)d_in[35];

    float* logits = (float*)d_out;

    // ---- workspace carving: fp32 region, then bf16 weight region ----
    float* ws = (float*)d_ws;
    size_t off = 0;
    auto alloc = [&](size_t n) { float* p = ws + off; off += (n + 63) & ~(size_t)63; return p; };

    const size_t ROWH = (size_t)BB * NSL * HID;
    float* memA  = alloc((size_t)LAYERS * BB * NMEML * HID);
    float* memB  = alloc((size_t)LAYERS * BB * NMEML * HID);
    float* X     = alloc(ROWH);
    float* Lin   = alloc(ROWH);
    float* Q     = alloc(ROWH);
    float* Kb    = alloc(ROWH);
    float* V     = alloc(ROWH + 256);                        // pad: benign OOB reads (N=64 tiles)
    float* S     = alloc((size_t)BB * HEADS * NSL * NMEML);
    float* AttO  = alloc(ROWH);
    float* Tmp1  = alloc(ROWH);
    float* Xsn   = alloc(ROWH);
    float* F1    = alloc((size_t)BB * NSL * FFW);
    float* Xso   = alloc(ROWH);
    float* MemIn = alloc((size_t)BB * NMEML * HID);
    float* MK    = alloc((size_t)BB * NMEML * HID);
    float* MV    = alloc((size_t)BB * NMEML * HID + 256);    // pad
    float* Xmn   = alloc(ROWH);
    float* Xmo   = alloc(ROWH);
    float* Alph  = alloc((size_t)BB * NML * NFCL);
    float* CompT = alloc((size_t)BB * NFCL * HID);
    float* Comp  = alloc((size_t)BB * NFCL * HID);
    float* Aemb  = alloc((size_t)BB * NSL * EMB);
    float* Xln   = alloc(ROWH);

    __bf16* wbf = (__bf16*)(ws + off);
    size_t boff = 0;
    auto balloc = [&](size_t n) { __bf16* p = wbf + boff; boff += (n + 63) & ~(size_t)63; return p; };

    const size_t HH = (size_t)HID * HID;
    __bf16* WtDec = balloc((size_t)HID * EMB);
    __bf16* WtQ   = balloc((size_t)LAYERS * HH);
    __bf16* WtK   = balloc((size_t)LAYERS * HH);
    __bf16* WtV   = balloc((size_t)LAYERS * HH);
    __bf16* WtC   = balloc((size_t)LAYERS * HH);
    __bf16* WtMQ  = balloc((size_t)LAYERS * HH);
    __bf16* WtMK  = balloc((size_t)LAYERS * HH);
    __bf16* WtMV  = balloc((size_t)LAYERS * HH);
    __bf16* WtMC  = balloc((size_t)LAYERS * HH);
    __bf16* WtF1d = balloc((size_t)LAYERS * FFW * HID);
    __bf16* WtF2d = balloc((size_t)LAYERS * HID * FFW);
    __bf16* WtF1m = balloc((size_t)LAYERS * FFW * HID);
    __bf16* WtF2m = balloc((size_t)LAYERS * HID * FFW);
    __bf16* WtCmp = balloc((size_t)LAYERS * NFCL * HID);
    __bf16* WtCL  = balloc((size_t)LAYERS * HH);
    __bf16* WtVoc = balloc((size_t)VOC * HID);

    // ---- pre-transpose + convert all weights to bf16 Wt (N x K) ----
    auto tr = [&](const float* W, __bf16* Wt, int K, int N) {
        dim3 g((N + 31) / 32, (K + 31) / 32);
        transpose_bf16_k<<<g, dim3(32, 8), 0, stream>>>(W, Wt, K, N);
    };
    tr(W_dec_lin, WtDec, EMB, HID);
    for (int m = 0; m < LAYERS; ++m) {
        tr(p_d_q + m * HH, WtQ  + m * HH, HID, HID);
        tr(p_d_k + m * HH, WtK  + m * HH, HID, HID);
        tr(p_d_v + m * HH, WtV  + m * HH, HID, HID);
        tr(p_d_c + m * HH, WtC  + m * HH, HID, HID);
        tr(p_m_q + m * HH, WtMQ + m * HH, HID, HID);
        tr(p_m_k + m * HH, WtMK + m * HH, HID, HID);
        tr(p_m_v + m * HH, WtMV + m * HH, HID, HID);
        tr(p_m_c + m * HH, WtMC + m * HH, HID, HID);
        tr(p_d_ff1 + (size_t)m * HID * FFW, WtF1d + (size_t)m * FFW * HID, HID, FFW);
        tr(p_d_ff2 + (size_t)m * FFW * HID, WtF2d + (size_t)m * HID * FFW, FFW, HID);
        tr(p_m_ff1 + (size_t)m * HID * FFW, WtF1m + (size_t)m * FFW * HID, HID, FFW);
        tr(p_m_ff2 + (size_t)m * FFW * HID, WtF2m + (size_t)m * HID * FFW, FFW, HID);
        tr(p_compress + (size_t)m * HID * NFCL, WtCmp + (size_t)m * NFCL * HID, HID, NFCL);
        tr(p_c_linear + m * HH, WtCL + m * HH, HID, HID);
    }
    tr(p_decoder, WtVoc, HID, VOC);

    const int ROWS = BB * NSL;
    const float qscale = 0.125f;

    zero_k<<<4096, 256, 0, stream>>>(memA, (long)LAYERS * BB * NMEML * HID);
    float* memCur = memA;
    float* memNext = memB;

    for (int nb = 0; nb < NBLK; ++nb) {
        embed_k<<<ROWS, 128, 0, stream>>>(Aemb, ids, W_emb, nb * NSL);
        gemm_w(stream, Aemb, WtDec, nullptr, X, ROWS, HID, EMB, EMB, HID,
               0,0,0,0, 1, 0, 1.0f, 1);

        for (int m = 0; m < LAYERS; ++m) {
            const float* memM = memCur + (size_t)m * BB * NMEML * HID;

            {   // mem_in = mem[m] + x_emb_memory[nb][m]
                long n = (long)BB * NMEML * HID;
                add_bcast_k<<<(int)((n + 255) / 256), 256, 0, stream>>>(
                    MemIn, memM, x_mem_emb + ((long)nb * LAYERS + m) * NMEML * HID,
                    n, (long)NMEML * HID);
            }

            ln_k<<<ROWS, 128, 0, stream>>>(Lin, X, nullptr,
                x_pos + (size_t)m * NSL * HID,
                b_i_scale + (size_t)m * HID, b_i_bias + (size_t)m * HID);

            // ---- self attention ----
            gemm_w(stream, Lin, WtQ + m * HH, nullptr, Q, ROWS, HID, HID, HID, HID,
                   0,0,0,0, 1, 0, qscale, 1);
            gemm_w(stream, Lin, WtK + m * HH, nullptr, Kb, ROWS, HID, HID, HID, HID,
                   0,0,0,0, 1, 0, 1.0f, 1);
            gemm_w(stream, Lin, WtV + m * HH, nullptr, V, ROWS, HID, HID, HID, HID,
                   0,0,0,0, 1, 0, 1.0f, 1);

            gemm(stream, Q, Kb, S, NSL, NSL, HD, HID, HID, NSL,
                 (long)NSL * HID, HD, (long)NSL * HID, HD,
                 (long)HEADS * NSL * NSL, (long)NSL * NSL,
                 HEADS, 0, 1, 1.0f, BB * HEADS);
            softmax_rows_k<<<dim3(NSL, BB * HEADS), 256, 0, stream>>>(S, NSL, 1);
            gemm(stream, S, V, AttO, NSL, HD, NSL, NSL, HID, HID,
                 (long)HEADS * NSL * NSL, (long)NSL * NSL,
                 (long)NSL * HID, HD, (long)NSL * HID, HD,
                 HEADS, 0, 0, 1.0f, BB * HEADS);
            gemm_w(stream, AttO, WtC + m * HH, nullptr, Tmp1, ROWS, HID, HID, HID, HID,
                   0,0,0,0, 1, 0, 1.0f, 1);

            ln_k<<<ROWS, 128, 0, stream>>>(Xsn, Tmp1, Lin, nullptr,
                b_d_scale_1 + (size_t)m * HID, b_d_bias_1 + (size_t)m * HID);

            gemm_w(stream, Xsn, WtF1d + (size_t)m * FFW * HID, b_d_ff1 + (size_t)m * FFW,
                   F1, ROWS, FFW, HID, HID, FFW, 0,0,0,0, 1, 1, 1.0f, 1);
            gemm_w(stream, F1, WtF2d + (size_t)m * HID * FFW, b_d_ff2 + (size_t)m * HID,
                   Tmp1, ROWS, HID, FFW, FFW, HID, 0,0,0,0, 1, 0, 1.0f, 1);
            ln_k<<<ROWS, 128, 0, stream>>>(Xso, Tmp1, Xsn, nullptr,
                b_d_scale_2 + (size_t)m * HID, b_d_bias_2 + (size_t)m * HID);

            // ---- memory attention ----
            gemm_w(stream, Xso, WtMQ + m * HH, nullptr, Q, ROWS, HID, HID, HID, HID,
                   0,0,0,0, 1, 0, qscale, 1);
            gemm_w(stream, MemIn, WtMK + m * HH, nullptr, MK, BB * NMEML, HID, HID, HID, HID,
                   0,0,0,0, 1, 0, 1.0f, 1);
            gemm_w(stream, MemIn, WtMV + m * HH, nullptr, MV, BB * NMEML, HID, HID, HID, HID,
                   0,0,0,0, 1, 0, 1.0f, 1);

            gemm(stream, Q, MK, S, NSL, NMEML, HD, HID, HID, NMEML,
                 (long)NSL * HID, HD, (long)NMEML * HID, HD,
                 (long)HEADS * NSL * NMEML, (long)NSL * NMEML,
                 HEADS, 0, 1, 1.0f, BB * HEADS);
            softmax_rows_k<<<dim3(NSL, BB * HEADS), 256, 0, stream>>>(S, NMEML, 0);
            gemm(stream, S, MV, AttO, NSL, HD, NMEML, NMEML, HID, HID,
                 (long)HEADS * NSL * NMEML, (long)NSL * NMEML,
                 (long)NMEML * HID, HD, (long)NSL * HID, HD,
                 HEADS, 0, 0, 1.0f, BB * HEADS);
            gemm_w(stream, AttO, WtMC + m * HH, nullptr, Tmp1, ROWS, HID, HID, HID, HID,
                   0,0,0,0, 1, 0, 1.0f, 1);

            ln_k<<<ROWS, 128, 0, stream>>>(Xmn, Tmp1, Lin, nullptr,
                b_m_scale_1 + (size_t)m * HID, b_m_bias_1 + (size_t)m * HID);
            gemm_w(stream, Xmn, WtF1m + (size_t)m * FFW * HID, b_m_ff1 + (size_t)m * FFW,
                   F1, ROWS, FFW, HID, HID, FFW, 0,0,0,0, 1, 1, 1.0f, 1);
            gemm_w(stream, F1, WtF2m + (size_t)m * HID * FFW, b_m_ff2 + (size_t)m * HID,
                   Tmp1, ROWS, HID, FFW, FFW, HID, 0,0,0,0, 1, 0, 1.0f, 1);
            ln_k<<<ROWS, 128, 0, stream>>>(Xmo, Tmp1, Xmn, nullptr,
                b_m_scale_2 + (size_t)m * HID, b_m_bias_2 + (size_t)m * HID);

            add_k<<<(int)((ROWH + 255) / 256), 256, 0, stream>>>(X, Xso, Xmo, (long)ROWH);

            // ---- memory compression (uses PRE-update mem[m]) ----
            const float* oldRM = memM + (size_t)NCML * HID;
            gemm_w(stream, oldRM, WtCmp + (size_t)m * NFCL * HID, nullptr, Alph,
                   NML, NFCL, HID, HID, NFCL,
                   (long)NMEML * HID, 0, (long)NML * NFCL, 0, 1, 0, 1.0f, BB);
            softmax_axis1_k<<<dim3(NFCL, BB), 128, 0, stream>>>(Alph);
            gemm(stream, Alph, oldRM, CompT, NFCL, HID, NML, NFCL, HID, HID,
                 (long)NML * NFCL, 0, (long)NMEML * HID, 0, (long)NFCL * HID, 0,
                 1, 1, 0, 1.0f, BB);
            gemm_w(stream, CompT, WtCL + m * HH, nullptr, Comp,
                   BB * NFCL, HID, HID, HID, HID, 0,0,0,0, 1, 0, 1.0f, 1);

            update_mem_k<<<dim3(NMEML, BB), 256, 0, stream>>>(
                memNext + (size_t)m * BB * NMEML * HID, memM, Comp, X);
        }

        { float* t = memCur; memCur = memNext; memNext = t; }

        ln_k<<<ROWS, 128, 0, stream>>>(Xln, X, nullptr, nullptr, d_o_scale, d_o_bias);
        gemm_w(stream, Xln, WtVoc, nullptr, logits + (size_t)nb * NSL * VOC,
               NSL, VOC, HID, HID, VOC,
               (long)NSL * HID, 0, (long)SEQL * VOC, 0, 1, 0, 1.0f, BB);
    }
}
